// DCGCNModel_9517647528299
// MI455X (gfx1250) — compile-verified, matrix operand order checked by reading, not run on hardware
//
#include <hip/hip_runtime.h>
#include <hip/hip_bf16.h>

// ---------------------------------------------------------------------------
// Types for CDNA5 WMMA (wave32): v16bf = 8 VGPRs of bf16 A/B fragment,
// v8f = 8 VGPRs of f32 C/D accumulator.
// ---------------------------------------------------------------------------
typedef __attribute__((ext_vector_type(16))) __bf16 v16bf;
typedef __attribute__((ext_vector_type(8)))  float  v8f;

union FragBF {
    v16bf        v;
    unsigned int u[8];
};

__device__ __forceinline__ unsigned short f2bf(float x) {
    unsigned u = __float_as_uint(x);
    unsigned r = u + 0x7FFFu + ((u >> 16) & 1u);   // round-to-nearest-even
    return (unsigned short)(r >> 16);
}

// Load one 16x32 bf16 fragment from a row-major tile (row stride = ld bf16
// elements). Per ISA 16-bit A/B layout (wave32): lane L holds row (L&15);
// half = L>>4; element j maps to k = k0 + half*8 + j + (j>=8?8:0), i.e.
// dwords [0..3] at (k0+half*8) and dwords [0..3] at (k0+half*8+16).
__device__ __forceinline__ void load_frag(FragBF& f, const unsigned short* base,
                                          int ld, int row, int k0, int lane) {
    const unsigned short* p = base + (size_t)row * ld + k0 + ((lane >> 4) << 3);
    const unsigned int* p32 = (const unsigned int*)p;
    f.u[0] = p32[0];  f.u[1] = p32[1];  f.u[2] = p32[2];  f.u[3] = p32[3];
    f.u[4] = p32[8];  f.u[5] = p32[9];  f.u[6] = p32[10]; f.u[7] = p32[11];
}

// ---------------------------------------------------------------------------
// CDNA5 async global->LDS copy (16B per lane), tracked by ASYNCcnt.
// lds_off: byte offset of destination within the wave's LDS allocation
// (low 32 bits of a generic pointer to __shared__ memory).
// ---------------------------------------------------------------------------
__device__ __forceinline__ void async_copy16(unsigned lds_off, const void* gaddr) {
#if defined(__HIP_DEVICE_COMPILE__)
    asm volatile("global_load_async_to_lds_b128 %0, %1, off"
                 :: "v"(lds_off), "v"(gaddr) : "memory");
#else
    (void)lds_off; (void)gaddr;
#endif
}

__device__ __forceinline__ void wait_async0() {
#if defined(__HIP_DEVICE_COMPILE__)
    asm volatile("s_wait_asynccnt 0" ::: "memory");
#endif
}

// ---------------------------------------------------------------------------
// bf16 GEMM with transposed-B operand, async-LDS double buffered:
//   D[b][m][n] = act( sum_k A[b][m][k] * Bt[n][k] + bias[n] )
// Block = 128 threads = 4 wave32s computing a 64x64 tile; K stepped by 64.
// Each step: block cooperatively issues global_load_async_to_lds_b128 for the
// NEXT 64-wide K-slice of A and Bt while computing v_wmma_f32_16x16x32_bf16
// on the current slice from LDS (ds_load_b128, conflict-free via 72-elem row
// stride). Requires M%64==0, N%64==0, K%64==0 (true for every GEMM here).
// Output f32 (Df) or bf16 (Dh) with arbitrary row/batch strides.
// ---------------------------------------------------------------------------
#define LDS_LD 72   // bf16 elems per LDS tile row (64 data + 8 pad) = 144 B

__global__ __launch_bounds__(128)
void gemm_bf16_bt_async(const unsigned short* __restrict__ A, long long sA, int ldA,
                        const unsigned short* __restrict__ Bt, long long sB, int ldB,
                        const float* __restrict__ bias,
                        float* __restrict__ Df, unsigned short* __restrict__ Dh,
                        long long sD, int ldD, int M, int Nn, int K, int act) {
    __shared__ __align__(16) unsigned short Abuf[2][64 * LDS_LD];
    __shared__ __align__(16) unsigned short Bbuf[2][64 * LDS_LD];

    const int b = blockIdx.z;
    const unsigned short* Ab = A  + (size_t)b * sA + (size_t)(blockIdx.y * 64) * ldA;
    const unsigned short* Bb = Bt + (size_t)b * sB + (size_t)(blockIdx.x * 64) * ldB;
    const int tid  = threadIdx.x;
    const int lane = tid & 31;
    const int w    = tid >> 5;

    // Issue the block's async copies for K-slice starting at k0 into buffer buf.
    // 64 rows x 128 B per matrix = 512 x 16B chunks; 128 threads x 4 chunks.
    auto issue = [&](int k0, int buf) {
        for (int q = 0; q < 4; ++q) {
            const int id  = tid + q * 128;
            const int row = id >> 3;        // 0..63
            const int c16 = id & 7;         // 16-byte chunk within the row
            const unsigned aoff =
                (unsigned)(size_t)&Abuf[buf][row * LDS_LD + c16 * 8];
            const unsigned boff =
                (unsigned)(size_t)&Bbuf[buf][row * LDS_LD + c16 * 8];
            async_copy16(aoff, Ab + (size_t)row * ldA + k0 + c16 * 8);
            async_copy16(boff, Bb + (size_t)row * ldB + k0 + c16 * 8);
        }
    };

    issue(0, 0);

    v8f acc[2][2] = {};
    const int m0w  = (w >> 1) * 32;         // wave tile within block
    const int n0w  = (w & 1) * 32;
    const int arow = m0w + (lane & 15);
    const int brow = n0w + (lane & 15);

    const int nsteps = K >> 6;
    for (int s = 0; s < nsteps; ++s) {
        wait_async0();          // this wave's async copies landed in LDS
        __syncthreads();        // every wave's copies landed; prev reads done
        if (s + 1 < nsteps) issue((s + 1) * 64, (s + 1) & 1);

        const unsigned short* At = Abuf[s & 1];
        const unsigned short* Bl = Bbuf[s & 1];
        for (int kk = 0; kk < 64; kk += 32) {
            FragBF a0, a1, b0, b1;
            load_frag(a0, At, LDS_LD, arow,      kk, lane);
            load_frag(a1, At, LDS_LD, arow + 16, kk, lane);
            load_frag(b0, Bl, LDS_LD, brow,      kk, lane);
            load_frag(b1, Bl, LDS_LD, brow + 16, kk, lane);
#if defined(__HIP_DEVICE_COMPILE__)
            acc[0][0] = __builtin_amdgcn_wmma_f32_16x16x32_bf16(
                false, a0.v, false, b0.v, (short)0, acc[0][0], false, false);
            acc[0][1] = __builtin_amdgcn_wmma_f32_16x16x32_bf16(
                false, a0.v, false, b1.v, (short)0, acc[0][1], false, false);
            acc[1][0] = __builtin_amdgcn_wmma_f32_16x16x32_bf16(
                false, a1.v, false, b0.v, (short)0, acc[1][0], false, false);
            acc[1][1] = __builtin_amdgcn_wmma_f32_16x16x32_bf16(
                false, a1.v, false, b1.v, (short)0, acc[1][1], false, false);
#else
            (void)a0; (void)a1; (void)b0; (void)b1;
#endif
        }
    }

    // C/D layout: lane holds col = (lane&15); VGPR r holds row = r + (lane>>4)*8
    const int m0   = blockIdx.y * 64 + m0w;
    const int n0   = blockIdx.x * 64 + n0w;
    const int colb   = lane & 15;
    const int rowoff = (lane >> 4) * 8;
    for (int i = 0; i < 2; ++i) {
        for (int j = 0; j < 2; ++j) {
            const int col  = n0 + j * 16 + colb;
            const float bv = bias ? bias[col] : 0.0f;
            for (int r = 0; r < 8; ++r) {
                const int row = m0 + i * 16 + rowoff + r;
                float v = acc[i][j][r] + bv;
                if (act == 1) v = fmaxf(v, 0.0f);
                const size_t idx = (size_t)b * sD + (size_t)row * ldD + col;
                if (Df) Df[idx] = v;
                else    Dh[idx] = f2bf(v);
            }
        }
    }
    (void)M; (void)Nn;
}

// ---------------------------------------------------------------------------
// f32 [batch,R,C] -> bf16 [batch,C,R]  (tiled transpose through LDS)
// ---------------------------------------------------------------------------
__global__ void transpose_cvt_k(const float* __restrict__ in,
                                unsigned short* __restrict__ out, int R, int C) {
    __shared__ float tile[32][33];
    const int b  = blockIdx.z;
    const int r0 = blockIdx.y * 32, c0 = blockIdx.x * 32;
    in  += (size_t)b * R * C;
    out += (size_t)b * R * C;
    const int tx = threadIdx.x, ty = threadIdx.y;
    for (int i = 0; i < 32; i += 8) {
        const int r = r0 + ty + i, c = c0 + tx;
        tile[ty + i][tx] = (r < R && c < C) ? in[(size_t)r * C + c] : 0.0f;
    }
    __syncthreads();
    for (int i = 0; i < 32; i += 8) {
        const int c = c0 + ty + i, r = r0 + tx;
        if (c < C && r < R) out[(size_t)c * R + r] = f2bf(tile[tx][ty + i]);
    }
}

// f32 strided [rows,cols] (row stride inLd) -> bf16 (row stride outLd)
__global__ void cvt2d_k(const float* __restrict__ in, int inLd,
                        unsigned short* __restrict__ out, int outLd,
                        long long rows, int cols) {
    long long i = blockIdx.x * (long long)blockDim.x + threadIdx.x;
    if (i >= rows * (long long)cols) return;
    long long r = i / cols;
    int c = (int)(i % cols);
    out[r * outLd + c] = f2bf(in[r * inLd + c]);
}

// ---------------------------------------------------------------------------
// Fused leaky-relu + adjacency mask + softmax over the last dim (Nn=1024);
// one 256-thread block per (b,n) row, bf16 output.
// ---------------------------------------------------------------------------
__global__ __launch_bounds__(256)
void softmax_mask_k(const float* __restrict__ scores, const int* __restrict__ adj,
                    unsigned short* __restrict__ attn, int Nn, unsigned maskbits) {
    __shared__ float red[256];
    const long long row = blockIdx.x;
    const float* s = scores + row * Nn;
    const int*   a = adj    + row * Nn;
    unsigned short* o = attn + row * Nn;
    const int t = threadIdx.x, per = Nn / 256;
    float vals[8];
    float lmax = -3.4e38f;
    for (int i = 0; i < per; ++i) {
        const int m = t + i * 256;
        float v = s[m];
        v = (v > 0.0f) ? v : 0.01f * v;              // leaky_relu (before mask)
        const int av = a[m];
        const bool ok = (av >= 0) && (av < 32) && ((maskbits >> av) & 1u);
        v = ok ? v : -1.0e9f;
        vals[i] = v;
        lmax = fmaxf(lmax, v);
    }
    red[t] = lmax; __syncthreads();
    for (int st = 128; st > 0; st >>= 1) {
        if (t < st) red[t] = fmaxf(red[t], red[t + st]);
        __syncthreads();
    }
    const float mx = red[0]; __syncthreads();
    float lsum = 0.0f;
    for (int i = 0; i < per; ++i) { vals[i] = __expf(vals[i] - mx); lsum += vals[i]; }
    red[t] = lsum; __syncthreads();
    for (int st = 128; st > 0; st >>= 1) {
        if (t < st) red[t] += red[t + st];
        __syncthreads();
    }
    const float inv = 1.0f / red[0];
    for (int i = 0; i < per; ++i) o[t + i * 256] = f2bf(vals[i] * inv);
}

// ---------------------------------------------------------------------------
// out = LayerNorm(z + resid) * g + b ;  one 128-thread block per row, C=384
// ---------------------------------------------------------------------------
__global__ __launch_bounds__(128)
void resid_ln_k(const float* __restrict__ z, const float* __restrict__ res,
                const float* __restrict__ g, const float* __restrict__ bb,
                float* __restrict__ out, int C) {
    __shared__ float red[128];
    const long long row = blockIdx.x;
    const float* zr = z + row * C;
    const float* rr = res + row * C;
    float* orow = out + row * C;
    const int t = threadIdx.x, per = C / 128;
    float v[4];
    float s = 0.0f;
    for (int i = 0; i < per; ++i) {
        const int c = t + i * 128;
        v[i] = zr[c] + rr[c];
        s += v[i];
    }
    red[t] = s; __syncthreads();
    for (int st = 64; st > 0; st >>= 1) { if (t < st) red[t] += red[t + st]; __syncthreads(); }
    const float mu = red[0] / C; __syncthreads();
    s = 0.0f;
    for (int i = 0; i < per; ++i) { const float d = v[i] - mu; s += d * d; }
    red[t] = s; __syncthreads();
    for (int st = 64; st > 0; st >>= 1) { if (t < st) red[t] += red[t + st]; __syncthreads(); }
    const float inv = rsqrtf(red[0] / C + 1e-5f);
    for (int i = 0; i < per; ++i) {
        const int c = t + i * 128;
        orow[c] = (v[i] - mu) * inv * g[c] + bb[c];
    }
}

// copy f32 [rows,cols] into wider output [rows,outLd] at column `off`
__global__ void copy_out_k(const float* __restrict__ in, float* __restrict__ out,
                           int cols, int outLd, int off) {
    const long long row = blockIdx.x;
    for (int c = threadIdx.x; c < cols; c += blockDim.x)
        out[row * (long long)outLd + off + c] = in[row * (long long)cols + c];
}

// ---------------------------------------------------------------------------
// Host orchestration
// ---------------------------------------------------------------------------
extern "C" void kernel_launch(void* const* d_in, const int* in_sizes, int n_in,
                              void* d_out, int out_size, void* d_ws, size_t ws_size,
                              hipStream_t stream) {
    (void)in_sizes; (void)n_in; (void)out_size; (void)ws_size;
    constexpr int Bb = 8, Nn = 1024, Dd = 768, DIM = 384;
    constexpr long long ROWS = (long long)Bb * Nn;   // 8192

    const int*   adj = (const int*)d_in[0];
    const float* h   = (const float*)d_in[1];

    // --- unpack params: jax dict-flatten order (sorted keys) --------------
    struct Layer {
        const float* attnW[6]; const float* fcb[6]; const float* fcw[6];
        const float *lfc_b, *lfc_w, *ln_b, *ln_g; int H;
    };
    struct Cell { const float *in_b, *in_w; Layer L[4]; };
    Cell cells[2];
    int p = 2;
    for (int c = 0; c < 2; ++c) {                    // 'directed' < 'reversed'
        cells[c].in_b = (const float*)d_in[p++];     // 'in_b' < 'in_w' < 'layers'
        cells[c].in_w = (const float*)d_in[p++];
        for (int l = 0; l < 4; ++l) {
            const int H = (l & 1) ? 3 : 6;
            Layer& ly = cells[c].L[l]; ly.H = H;
            for (int i = 0; i < H; ++i) ly.attnW[i] = (const float*)d_in[p++];
            for (int i = 0; i < H; ++i) ly.fcb[i]   = (const float*)d_in[p++];
            for (int i = 0; i < H; ++i) ly.fcw[i]   = (const float*)d_in[p++];
            ly.lfc_b = (const float*)d_in[p++];
            ly.lfc_w = (const float*)d_in[p++];
            ly.ln_b  = (const float*)d_in[p++];
            ly.ln_g  = (const float*)d_in[p++];
        }
    }

    // --- workspace ---------------------------------------------------------
    char* wsp = (char*)d_ws;
    auto alloc = [&](size_t bytes) -> char* {
        char* r = wsp; wsp += (bytes + 255) / 256 * 256; return r;
    };
    unsigned short* h_in_bf = (unsigned short*)alloc(ROWS * Dd  * 2);
    unsigned short* hcat_bf = (unsigned short*)alloc(ROWS * Dd  * 2);
    float*          hcur    = (float*)         alloc(ROWS * DIM * 4);
    float*          hnext   = (float*)         alloc(ROWS * DIM * 4);
    float*          xf      = (float*)         alloc(ROWS * 128 * 4);
    unsigned short* xbf     = (unsigned short*)alloc(ROWS * 128 * 2);
    unsigned short* ybf     = (unsigned short*)alloc(ROWS * 128 * 2);
    unsigned short* xT      = (unsigned short*)alloc((size_t)Bb * 128 * Nn * 2);
    float*          scores  = (float*)         alloc((size_t)Bb * Nn * Nn * 4);
    unsigned short* attnbf  = (unsigned short*)alloc((size_t)Bb * Nn * Nn * 2);
    float*          zbuf    = (float*)         alloc(ROWS * DIM * 4);
    unsigned short* wt0     = (unsigned short*)alloc((size_t)Dd * DIM * 2);
    unsigned short* wt1     = (unsigned short*)alloc((size_t)128 * 128 * 2);

    auto gemm = [&](const unsigned short* A, long long sA, int ldA,
                    const unsigned short* Bt, long long sB, int ldB,
                    const float* bias, float* Df, unsigned short* Dh,
                    long long sD, int ldD, int M, int Np, int K, int act, int batch) {
        dim3 g((Np + 63) / 64, (M + 63) / 64, batch);
        hipLaunchKernelGGL(gemm_bf16_bt_async, g, dim3(128), 0, stream,
                           A, sA, ldA, Bt, sB, ldB, bias, Df, Dh, sD, ldD, M, Np, K, act);
    };
    auto transp = [&](const float* in, unsigned short* out, int R, int C, int batch) {
        dim3 g((C + 31) / 32, (R + 31) / 32, batch);
        hipLaunchKernelGGL(transpose_cvt_k, g, dim3(32, 8), 0, stream, in, out, R, C);
    };
    auto cvt2d = [&](const float* in, int inLd, unsigned short* out, int outLd,
                     long long rows, int cols) {
        long long tot = rows * cols;
        hipLaunchKernelGGL(cvt2d_k, dim3((unsigned)((tot + 255) / 256)), dim3(256),
                           0, stream, in, inLd, out, outLd, rows, cols);
    };

    // original h -> bf16 once (A operand for both cells' input GEMM)
    cvt2d(h, Dd, h_in_bf, Dd, ROWS, Dd);

    for (int c = 0; c < 2; ++c) {
        const unsigned maskbits = (c == 0) ? ((1u << 1) | (1u << 3) | (1u << 4))
                                           : ((1u << 2) | (1u << 3) | (1u << 4));
        const Cell& cp = cells[c];

        // h0 = h @ in_w + in_b   (Bt = in_w^T [384,768])
        transp(cp.in_w, wt0, Dd, DIM, 1);
        gemm(h_in_bf, 0, Dd, wt0, 0, Dd, cp.in_b, hcur, nullptr,
             0, DIM, (int)ROWS, DIM, Dd, 0, 1);

        for (int l = 0; l < 4; ++l) {
            const Layer& ly = cp.L[l];
            const int hid = DIM / ly.H;

            // seed concat buffer cols [0,384) with layer input
            cvt2d(hcur, DIM, hcat_bf, Dd, ROWS, DIM);

            for (int i = 0; i < ly.H; ++i) {
                const int din = DIM + hid * i;    // always a multiple of 64
                // x = hcat[:, :din] @ fc_w + fc_b
                transp(ly.fcw[i], wt0, din, hid, 1);
                gemm(hcat_bf, 0, Dd, wt0, 0, din, ly.fcb[i], xf, nullptr,
                     0, hid, (int)ROWS, hid, din, 0, 1);
                cvt2d(xf, hid, xbf, hid, ROWS, hid);          // x  -> bf16
                transp(xf, xT, Nn, hid, Bb);                  // x^T -> bf16 [B,hid,N]
                // y = x @ attn_W
                transp(ly.attnW[i], wt1, hid, hid, 1);
                gemm(xbf, 0, hid, wt1, 0, hid, nullptr, nullptr, ybf,
                     0, hid, (int)ROWS, hid, hid, 0, 1);
                // scores = y @ x^T  (Bt operand is x row-major)
                gemm(ybf, (long long)Nn * hid, hid, xbf, (long long)Nn * hid, hid,
                     nullptr, scores, nullptr, (long long)Nn * Nn, Nn,
                     Nn, Nn, hid, 0, Bb);
                // leaky-relu + mask + softmax -> bf16
                hipLaunchKernelGGL(softmax_mask_k, dim3((unsigned)ROWS), dim3(256),
                                   0, stream, scores, adj, attnbf, Nn, maskbits);
                // c_i = relu(attn @ x), written bf16 straight into concat block
                gemm(attnbf, (long long)Nn * Nn, Nn, xT, (long long)hid * Nn, Nn,
                     nullptr, nullptr, hcat_bf + DIM + hid * i,
                     (long long)Nn * Dd, Dd, Nn, hid, Nn, 1, Bb);
            }

            // z = relu(concat(heads) @ layer_fc_w + b) ; then LN(z + input)
            transp(ly.lfc_w, wt0, DIM, DIM, 1);
            gemm(hcat_bf + DIM, 0, Dd, wt0, 0, DIM, ly.lfc_b, zbuf, nullptr,
                 0, DIM, (int)ROWS, DIM, DIM, 1, 1);
            hipLaunchKernelGGL(resid_ln_k, dim3((unsigned)ROWS), dim3(128),
                               0, stream, zbuf, hcur, ly.ln_g, ly.ln_b, hnext, DIM);
            float* tmp = hcur; hcur = hnext; hnext = tmp;
        }

        hipLaunchKernelGGL(copy_out_k, dim3((unsigned)ROWS), dim3(256), 0, stream,
                           hcur, (float*)d_out, DIM, Dd, c * DIM);
    }
}
